// DyvitPruner_73272142069915
// MI455X (gfx1250) — matrix operand order repaired and checked by reading, not compiled, not run on hardware
//
#include <hip/hip_runtime.h>
#include <hip/hip_bf16.h>
#include <math.h>
#include <stddef.h>
#include <stdint.h>

// ---------------------------------------------------------------------------
// Types for WMMA fragments (CDNA5 / gfx1250, wave32)
// ---------------------------------------------------------------------------
typedef __attribute__((ext_vector_type(16))) __bf16 v16bf;
typedef __attribute__((ext_vector_type(8)))  __bf16 v8bf;
typedef __attribute__((ext_vector_type(4)))  __bf16 v4bf;
typedef __attribute__((ext_vector_type(8)))  float  v8f;
typedef __attribute__((ext_vector_type(4)))  float  v4f;

#define BATCH  64
#define NIMG   577
#define NTOK   576          // NIMG - 1
#define DMODEL 1024
#define MROWS  (BATCH * NTOK)   // 36864
#define NKEEP  403              // int(577 * 0.7)

__device__ __forceinline__ float gelu_exact(float x) {
  return 0.5f * x * (1.0f + erff(x * 0.70710678118654752440f));
}

// ---------------------------------------------------------------------------
// gfx1250 async global->LDS copy (ISA 15.18.3 op 98, ASYNCcnt-tracked).
// GVS addressing: 64-bit SGPR base + 32-bit per-lane byte offset.
// dst operand is the per-lane LDS byte offset (low 32 bits of a generic
// pointer to __shared__ memory -- LDS aperture maps addr[31:0] to the
// workgroup's LDS allocation).
// ---------------------------------------------------------------------------
__device__ __forceinline__ void async_copy_b128(uint32_t lds_byte,
                                                unsigned long long gbase,
                                                uint32_t goff) {
  asm volatile("global_load_async_to_lds_b128 %0, %1, %2"
               :: "v"(lds_byte), "v"(goff), "s"(gbase)
               : "memory");
}
__device__ __forceinline__ void wait_async_le(int prev_in_flight) {
  if (prev_in_flight)
    asm volatile("s_wait_asynccnt 0x4" ::: "memory");
  else
    asm volatile("s_wait_asynccnt 0x0" ::: "memory");
}

// ---------------------------------------------------------------------------
// fused transpose + float->bf16 cast for weights:
//   in:  K x N row-major f32,  out: N x K row-major bf16  (= B column-major)
// so GEMM B-tiles load with pure vector transfers, no LDS transpose.
// ---------------------------------------------------------------------------
__global__ __launch_bounds__(256)
void f2bf_transpose_kernel(const float* __restrict__ in,
                           __bf16* __restrict__ out, int K, int N) {
  __shared__ __bf16 tile[32][33];
  const int k0 = blockIdx.y * 32;
  const int n0 = blockIdx.x * 32;
  const int tx = threadIdx.x & 31;
  const int ty = threadIdx.x >> 5;   // 0..7
#pragma unroll
  for (int p = 0; p < 4; ++p) {
    const int k = ty + p * 8;
    tile[k][tx] = (__bf16)in[(size_t)(k0 + k) * N + n0 + tx];
  }
  __syncthreads();
#pragma unroll
  for (int p = 0; p < 4; ++p) {
    const int n = ty + p * 8;
    out[(size_t)(n0 + n) * K + k0 + tx] = tile[tx][n];
  }
}

// ---------------------------------------------------------------------------
// LayerNorm over D=1024 of image tokens 1..576, output bf16 rows
// one block (256 thr) per row
// ---------------------------------------------------------------------------
__global__ __launch_bounds__(256)
void ln_cast_kernel(const float* __restrict__ img,   // (B,577,1024)
                    const float* __restrict__ g,
                    const float* __restrict__ beta,
                    __bf16* __restrict__ out) {      // (MROWS,1024)
  __shared__ float redS[8], redS2[8];
  const int row  = blockIdx.x;
  const int bidx = row / NTOK;
  const int tok  = row - bidx * NTOK;
  const float* x = img + ((size_t)bidx * NIMG + 1 + tok) * DMODEL;
  const int t = threadIdx.x;

  v4f v = ((const v4f*)x)[t];
  float s  = v.x + v.y + v.z + v.w;
  float s2 = v.x * v.x + v.y * v.y + v.z * v.z + v.w * v.w;
#pragma unroll
  for (int o = 16; o > 0; o >>= 1) {
    s  += __shfl_xor(s, o, 32);
    s2 += __shfl_xor(s2, o, 32);
  }
  const int wid = t >> 5, lid = t & 31;
  if (lid == 0) { redS[wid] = s; redS2[wid] = s2; }
  __syncthreads();
  float S = 0.f, S2 = 0.f;
#pragma unroll
  for (int i = 0; i < 8; ++i) { S += redS[i]; S2 += redS2[i]; }

  const float mu   = S * (1.0f / DMODEL);
  const float var  = S2 * (1.0f / DMODEL) - mu * mu;
  const float rinv = rsqrtf(var + 1e-5f);

  v4f gg = ((const v4f*)g)[t];
  v4f bb = ((const v4f*)beta)[t];
  v4bf o;
  o[0] = (__bf16)((v.x - mu) * rinv * gg.x + bb.x);
  o[1] = (__bf16)((v.y - mu) * rinv * gg.y + bb.y);
  o[2] = (__bf16)((v.z - mu) * rinv * gg.z + bb.z);
  o[3] = (__bf16)((v.w - mu) * rinv * gg.w + bb.w);
  ((v4bf*)(out + (size_t)row * DMODEL))[t] = o;
}

// ---------------------------------------------------------------------------
// bf16 WMMA GEMM: C = gelu(A @ B + bias)
//   A : M x K row-major bf16
//   Bt: N x K row-major bf16 (pre-transposed B)
//   Block tile 128x128x32, 256 threads = 8 waves (4 x 2), wave tile 32x64.
//   Double-buffered LDS filled by global_load_async_to_lds_b128; WMMAs on
//   the front buffer overlap the async DMA into the back buffer.
// ---------------------------------------------------------------------------
template <int OUT_BF16>
__global__ __launch_bounds__(256)
void gemm_bf16_gelu(const __bf16* __restrict__ A,
                    const __bf16* __restrict__ Bt,
                    const float* __restrict__ bias,
                    void* __restrict__ Cout,
                    int M, int Nn, int K) {
  constexpr int BM = 128, BN = 128, BK = 32;
  constexpr int LDA = BK + 8;   // bf16 units; pad keeps 16B alignment
  constexpr int LDB = BK + 8;
  __shared__ alignas(16) __bf16 sA[2][BM * LDA];
  __shared__ alignas(16) __bf16 sB[2][BN * LDB];

  const int t  = threadIdx.x;
  const int m0 = blockIdx.y * BM;
  const int n0 = blockIdx.x * BN;

  // tile loader mapping: 2 threads per row, 16 bf16 (2 x b128) each
  const int rowT = t >> 1;
  const int khT  = (t & 1) * 16;

  const int wave = t >> 5;
  const int lane = t & 31;
  const int wm = wave & 3;      // 0..3 -> row group of 32
  const int wn = wave >> 2;     // 0..1 -> col group of 64
  const int lr = lane & 15;
  const int lh = lane >> 4;

  const unsigned long long baseA = (unsigned long long)(uintptr_t)A;
  const unsigned long long baseB = (unsigned long long)(uintptr_t)Bt;

  v8f acc[2][4] = {};
  const int nk = K / BK;

  auto issue_tile = [&](int buf, int k0) {
    const uint32_t aoff = (uint32_t)((((size_t)(m0 + rowT)) * K + k0 + khT) * 2);
    const uint32_t alds = (uint32_t)(uintptr_t)&sA[buf][rowT * LDA + khT];
    async_copy_b128(alds,      baseA, aoff);
    async_copy_b128(alds + 16, baseA, aoff + 16);
    const uint32_t boff = (uint32_t)((((size_t)(n0 + rowT)) * K + k0 + khT) * 2);
    const uint32_t blds = (uint32_t)(uintptr_t)&sB[buf][rowT * LDB + khT];
    async_copy_b128(blds,      baseB, boff);
    async_copy_b128(blds + 16, baseB, boff + 16);
  };

  issue_tile(0, 0);

  union FragU { v16bf v; v8bf h[2]; };

  for (int kt = 0; kt < nk; ++kt) {
    const int buf   = kt & 1;
    const bool more = (kt + 1) < nk;
    if (more) issue_tile(buf ^ 1, (kt + 1) * BK);   // DMA next slab -> back buffer
    wait_async_le(more);                            // retire previous slab's 4 copies
    __syncthreads();                                // publish front buffer

    // --- fragments from LDS per ISA layouts (CDNA5 ISA §7.12.2) ---
    FragU fa[2], fb[4];
#pragma unroll
    for (int i = 0; i < 2; ++i) {
      // A 16x32: lane<16 -> M=lane, K 0..7 & 16..23; lane>=16 -> K 8..15 & 24..31
      const __bf16* p = &sA[buf][(wm * 32 + i * 16 + lr) * LDA + lh * 8];
      fa[i].h[0] = *(const v8bf*)p;
      fa[i].h[1] = *(const v8bf*)(p + 16);
    }
#pragma unroll
    for (int j = 0; j < 4; ++j) {
      // B 32x16: lane<16 -> N=lane, K 0..7 & 16..23; lane>=16 -> K 8..15 & 24..31 (sB is col-major of B)
      const __bf16* p = &sB[buf][(wn * 64 + j * 16 + lr) * LDB + lh * 8];
      fb[j].h[0] = *(const v8bf*)p;
      fb[j].h[1] = *(const v8bf*)(p + 16);
    }

#pragma unroll
    for (int i = 0; i < 2; ++i)
#pragma unroll
      for (int j = 0; j < 4; ++j)
        acc[i][j] = __builtin_amdgcn_wmma_f32_16x16x32_bf16(
            false, fa[i].v, false, fb[j].v, (short)0, acc[i][j], false, false);

    __syncthreads();   // all waves done reading front buffer before it is refilled
  }

  // epilogue: C 16x16 f32 layout -> row = lh*8 + r, col = lr
#pragma unroll
  for (int i = 0; i < 2; ++i) {
#pragma unroll
    for (int j = 0; j < 4; ++j) {
      const int col  = n0 + wn * 64 + j * 16 + lr;
      const float bv = bias[col];
#pragma unroll
      for (int r = 0; r < 8; ++r) {
        const int row = m0 + wm * 32 + i * 16 + lh * 8 + r;
        float v = acc[i][j][r] + bv;
        v = gelu_exact(v);
        if (OUT_BF16)
          ((__bf16*)Cout)[(size_t)row * Nn + col] = (__bf16)v;
        else
          ((float*)Cout)[(size_t)row * Nn + col] = v;
      }
    }
  }
}

// ---------------------------------------------------------------------------
// Head: logits = h3 @ w4 + b4 (256 -> 2), log_softmax, keep class-0 score.
// One wave per row, 8 rows per block.
// ---------------------------------------------------------------------------
__global__ __launch_bounds__(256)
void scores_kernel(const float* __restrict__ h3,   // MROWS x 256
                   const float* __restrict__ w4,   // 256 x 2 row-major
                   const float* __restrict__ b4,
                   float* __restrict__ scores) {   // MROWS
  const int row  = blockIdx.x * 8 + (threadIdx.x >> 5);
  const int lane = threadIdx.x & 31;
  const float* h = h3 + (size_t)row * 256;
  float l0 = 0.f, l1 = 0.f;
#pragma unroll
  for (int k = lane; k < 256; k += 32) {
    float hv = h[k];
    l0 += hv * w4[2 * k];
    l1 += hv * w4[2 * k + 1];
  }
#pragma unroll
  for (int o = 16; o > 0; o >>= 1) {
    l0 += __shfl_xor(l0, o, 32);
    l1 += __shfl_xor(l1, o, 32);
  }
  if (lane == 0) {
    l0 += b4[0];
    l1 += b4[1];
    float m   = fmaxf(l0, l1);
    float lse = m + logf(expf(l0 - m) + expf(l1 - m));
    scores[row] = l0 - lse;
  }
}

// ---------------------------------------------------------------------------
// Top-k via stable rank (matches lax.top_k: descending, ties by lower index)
// one block per batch, 576 threads
// ---------------------------------------------------------------------------
__global__ __launch_bounds__(576)
void topk_kernel(const float* __restrict__ scores,  // B x 576
                 int* __restrict__ keep_idx_i,      // B x 403 (ws)
                 float* __restrict__ keep_idx_f,    // B x 403 (out)
                 float* __restrict__ topk_vals) {   // B x 403 (out)
  __shared__ float s[NTOK];
  const int b = blockIdx.x;
  const int t = threadIdx.x;
  s[t] = scores[(size_t)b * NTOK + t];
  __syncthreads();
  const float mine = s[t];
  int rank = 0;
  for (int j = 0; j < NTOK; ++j) {
    float o = s[j];
    rank += (o > mine) || (o == mine && j < t);
  }
  if (rank < NKEEP) {
    keep_idx_i[b * NKEEP + rank] = t;
    keep_idx_f[b * NKEEP + rank] = (float)t;
    topk_vals[b * NKEEP + rank]  = mine;
  }
}

// ---------------------------------------------------------------------------
// Gather: out row 0 = cls token, rows 1..403 = kept tokens
// ---------------------------------------------------------------------------
__global__ __launch_bounds__(256)
void gather_kernel(const float* __restrict__ img,   // B x 577 x 1024
                   const int* __restrict__ keep_idx,
                   float* __restrict__ out) {       // B x 404 x 1024
  const int r = blockIdx.x;   // 0..403
  const int b = blockIdx.y;
  const int src_tok = (r == 0) ? 0 : (1 + keep_idx[b * NKEEP + r - 1]);
  const v4f* src = (const v4f*)(img + ((size_t)b * NIMG + src_tok) * DMODEL);
  v4f* dst = (v4f*)(out + ((size_t)b * (NKEEP + 1) + r) * DMODEL);
  dst[threadIdx.x] = src[threadIdx.x];
}

__global__ void fill_ones_kernel(float* __restrict__ p, int n) {
  int i = blockIdx.x * blockDim.x + threadIdx.x;
  if (i < n) p[i] = 1.0f;
}

// ---------------------------------------------------------------------------
// Host launcher
// ---------------------------------------------------------------------------
extern "C" void kernel_launch(void* const* d_in, const int* in_sizes, int n_in,
                              void* d_out, int out_size, void* d_ws, size_t ws_size,
                              hipStream_t stream) {
  (void)in_sizes; (void)n_in; (void)out_size; (void)ws_size;

  const float* image_states = (const float*)d_in[3];
  const float* ln_g = (const float*)d_in[7];
  const float* ln_b = (const float*)d_in[8];
  const float* w1   = (const float*)d_in[9];
  const float* b1   = (const float*)d_in[10];
  const float* w2   = (const float*)d_in[11];
  const float* b2   = (const float*)d_in[12];
  const float* w3   = (const float*)d_in[13];
  const float* b3   = (const float*)d_in[14];
  const float* w4   = (const float*)d_in[15];
  const float* b4   = (const float*)d_in[16];

  // workspace layout
  char* ws = (char*)d_ws;
  size_t off = 0;
  auto wsalloc = [&](size_t bytes) -> char* {
    char* p = ws + off;
    off += (bytes + 255) & ~(size_t)255;
    return p;
  };
  __bf16* w1t  = (__bf16*)wsalloc((size_t)1024 * 1024 * 2);  // (N,K) bf16
  __bf16* w2t  = (__bf16*)wsalloc((size_t)512 * 1024 * 2);
  __bf16* w3t  = (__bf16*)wsalloc((size_t)256 * 512 * 2);
  __bf16* xln  = (__bf16*)wsalloc((size_t)MROWS * 1024 * 2);
  __bf16* h1   = (__bf16*)wsalloc((size_t)MROWS * 1024 * 2);
  int*    kidx = (int*)  wsalloc((size_t)BATCH * NKEEP * 4);
  __bf16* h2 = xln;          // xln dead after GEMM1
  float*  h3 = (float*)h1;   // h1 dead after GEMM2; f32 fits in bf16 region x2

  // output layout (flat, return order, all float)
  float* out      = (float*)d_out;
  float* o_states = out;
  float* o_mask   = o_states + (size_t)BATCH * (NKEEP + 1) * DMODEL;
  float* o_kidx   = o_mask + BATCH * (NKEEP + 1);
  float* o_scores = o_kidx + BATCH * NKEEP;
  float* o_topk   = o_scores + BATCH * NTOK;

  // 1) transpose + cast weights to bf16 (B becomes column-major => vector loads)
  f2bf_transpose_kernel<<<dim3(1024 / 32, 1024 / 32), 256, 0, stream>>>(w1, w1t, 1024, 1024);
  f2bf_transpose_kernel<<<dim3(512 / 32, 1024 / 32), 256, 0, stream>>>(w2, w2t, 1024, 512);
  f2bf_transpose_kernel<<<dim3(256 / 32, 512 / 32), 256, 0, stream>>>(w3, w3t, 512, 256);

  // 2) layernorm + cast
  ln_cast_kernel<<<MROWS, 256, 0, stream>>>(image_states, ln_g, ln_b, xln);

  // 3) three WMMA GEMMs with fused bias + exact GELU (async-LDS pipelined)
  gemm_bf16_gelu<1><<<dim3(1024 / 128, MROWS / 128), 256, 0, stream>>>(
      xln, w1t, b1, h1, MROWS, 1024, 1024);
  gemm_bf16_gelu<1><<<dim3(512 / 128, MROWS / 128), 256, 0, stream>>>(
      h1, w2t, b2, h2, MROWS, 512, 1024);
  gemm_bf16_gelu<0><<<dim3(256 / 128, MROWS / 128), 256, 0, stream>>>(
      h2, w3t, b3, h3, MROWS, 256, 512);

  // 4) head + log_softmax -> scores (written straight into d_out)
  scores_kernel<<<MROWS / 8, 256, 0, stream>>>(h3, w4, b4, o_scores);

  // 5) stable-rank top-k per batch
  topk_kernel<<<BATCH, NTOK, 0, stream>>>(o_scores, kidx, o_kidx, o_topk);

  // 6) gather kept tokens (+ cls) and write mask
  gather_kernel<<<dim3(NKEEP + 1, BATCH), 256, 0, stream>>>(image_states, kidx, o_states);
  fill_ones_kernel<<<(BATCH * (NKEEP + 1) + 255) / 256, 256, 0, stream>>>(
      o_mask, BATCH * (NKEEP + 1));
}
